// HierarchicalVQVAE_61710090109543
// MI455X (gfx1250) — matrix operand order repaired and checked by reading, not compiled, not run on hardware
//
#include <hip/hip_runtime.h>
#include <hip/hip_bf16.h>
#include <stdint.h>

typedef __attribute__((ext_vector_type(16))) __bf16 v16bf;
typedef __attribute__((ext_vector_type(8)))  float  v8f;

#define D_CH   512
#define T_LEN  512
#define B_SZ   32
#define M_TOT  (B_SZ * T_LEN)          // 16384
#define NLAY   20
#define KZ_N   128
#define KQ_N   64
#define A_DIM  14
#define WPERL  786432                  // 512*512*3 elements per layer

__device__ __forceinline__ unsigned short f2bf(float f) {
    unsigned int u = __float_as_uint(f);
    unsigned int r = u + 0x7FFFu + ((u >> 16) & 1u);
    return (unsigned short)(r >> 16);
}

// ---------------------------------------------------------------------------
// Weight prep: fp32 [L][O][I][3] -> bf16 WMMA B-fragment order
// flat = ((c*32 + nb)*16 + kb)*512 + lane*16 + h
//   element = w[o = nb*16 + lane%16][i = kb*32 + h + 16*(lane/16)][c]
// ---------------------------------------------------------------------------
__global__ __launch_bounds__(256) void k_wprep(const float* __restrict__ we,
                                               const float* __restrict__ wd,
                                               unsigned short* __restrict__ dst) {
    const size_t id   = (size_t)blockIdx.x * 256 + threadIdx.x;
    const size_t HALF = (size_t)WPERL * NLAY;
    const int    set  = (int)(id / HALF);
    const size_t rem  = id - (size_t)set * HALF;
    const int l  = (int)(rem / WPERL);
    const int e  = (int)(rem - (size_t)l * WPERL);
    const int c  = e >> 18;
    const int r1 = e & 262143;
    const int nb = r1 >> 13;
    const int r2 = r1 & 8191;
    const int kb = r2 >> 9;
    const int q  = r2 & 511;
    const int ln = q >> 4;
    const int h  = q & 15;
    const int o  = nb * 16 + (ln & 15);
    const int i  = kb * 32 + h + ((ln >> 4) << 4);
    const float* src = set ? wd : we;
    const float v = src[(((size_t)l * 512 + o) * 512 + i) * 3 + c];
    dst[id] = f2bf(v);
}

// ---------------------------------------------------------------------------
// Encoder input projection: [M,14] @ [14,512] + b -> fp32 + bf16 mirrors
// ---------------------------------------------------------------------------
__global__ __launch_bounds__(256) void k_proj_in(const float* __restrict__ act,
                                                 const float* __restrict__ W,
                                                 const float* __restrict__ b,
                                                 float* __restrict__ xf,
                                                 unsigned short* __restrict__ xh) {
    const int m = blockIdx.x >> 1;
    const int d = ((blockIdx.x & 1) << 8) + threadIdx.x;
    const float* a = act + (size_t)m * A_DIM;
    float acc = b[d];
#pragma unroll
    for (int i = 0; i < A_DIM; ++i) acc += a[i] * W[i * D_CH + d];
    const size_t off = (size_t)m * D_CH + d;
    xf[off] = acc;
    xh[off] = f2bf(acc);
}

// ---------------------------------------------------------------------------
// One TCN layer: y = relu(dilated_conv3(x) + bias) + x, via bf16 WMMA.
// Block: 256 thr (8 waves), tile 128(M) x 128(N).
// K-loop: 48 steps (3 taps x 16 x K32), double-buffered LDS A-tile staged via
// GLOBAL_LOAD_ASYNC_TO_LDS_B128 (ASYNCcnt), B fragments straight from L2.
// ---------------------------------------------------------------------------
__global__ __launch_bounds__(256) void k_tcn(const float* __restrict__ xf,
                                             const unsigned short* __restrict__ xh,
                                             float* __restrict__ yf,
                                             unsigned short* __restrict__ yh,
                                             const unsigned short* __restrict__ wfrag,
                                             const float* __restrict__ bias,
                                             int dil) {
    __shared__ unsigned short lds_a[2][128][40];   // pitch 40 halves: conflict-free
    const int tid  = threadIdx.x;
    const int lane = tid & 31;
    const int wid  = tid >> 5;
    const int wm   = wid & 3;                      // 4 waves along M
    const int wn   = wid >> 2;                     // 2 waves along N
    const int m0   = blockIdx.x * 128;
    const int n0   = blockIdx.y * 128;
    const int g    = lane >> 4;
    const int nl   = lane & 15;

    v8f acc[2][4];
#pragma unroll
    for (int i = 0; i < 2; ++i)
#pragma unroll
        for (int j = 0; j < 4; ++j)
            acc[i][j] = v8f{0.f, 0.f, 0.f, 0.f, 0.f, 0.f, 0.f, 0.f};

    const int ar     = tid >> 1;                   // A staging: row 0..127
    const int acol   = (tid & 1) << 4;             // half-col 0 or 16
    const int nbBase = blockIdx.y * 8 + wn * 4;
    const int tbase  = (m0 + ar) & (T_LEN - 1);

    const unsigned lds0 = (unsigned)(uintptr_t)&lds_a[0][ar][acol];
    const unsigned lds1 = (unsigned)(uintptr_t)&lds_a[1][ar][acol];

    // async-stage one 128x32 bf16 A tile (tap c = step/16, k-block = step%16)
    auto stage = [&](int step, int buf) {
        const int c     = step >> 4;
        const int kb    = step & 15;
        const int shift = (c - 1) * dil;
        if ((unsigned)(tbase + shift) < (unsigned)T_LEN) {
            const unsigned short* gp =
                xh + (ptrdiff_t)(m0 + ar + shift) * D_CH + kb * 32 + acol;
            const unsigned long long ga = (unsigned long long)(uintptr_t)gp;
            const unsigned la = buf ? lds1 : lds0;
            asm volatile(
                "global_load_async_to_lds_b128 %0, %1, off\n\t"
                "global_load_async_to_lds_b128 %0, %1, off offset:16"
                :: "v"(la), "v"(ga) : "memory");
        } else {
            uint4* ad = reinterpret_cast<uint4*>(&lds_a[buf][ar][acol]);
            const uint4 z = make_uint4(0u, 0u, 0u, 0u);
            ad[0] = z;
            ad[1] = z;
        }
    };

    stage(0, 0);
#pragma unroll 2
    for (int step = 0; step < 48; ++step) {
        const int buf = step & 1;
        // drain our async copies into lds_a[buf], then make them WG-visible
        asm volatile("s_wait_asynccnt 0x0" ::: "memory");
        __syncthreads();
        // kick off next tile's DMA under this step's WMMAs
        if (step + 1 < 48) stage(step + 1, buf ^ 1);

        const int c  = step >> 4;
        const int kb = step & 15;

        // ---- B fragments straight from global (L2-resident, pre-shuffled) ----
        union UB { v16bf v; uint4 q[2]; } bu[4];
        const uint4* wp = reinterpret_cast<const uint4*>(
            wfrag + ((((size_t)c * 32 + nbBase) * 16 + kb) * 512) + lane * 16);
#pragma unroll
        for (int nt = 0; nt < 4; ++nt) {
            const uint4* p = wp + nt * 1024;       // nb stride = 8192 halves
            bu[nt].q[0] = p[0];
            bu[nt].q[1] = p[1];
        }

        // ---- A fragments from LDS (contiguous k = 16*(lane/16) + s, same as B) ----
        union UA { v16bf v; unsigned int u[8]; } au[2];
#pragma unroll
        for (int mi = 0; mi < 2; ++mi) {
            const int row = wm * 32 + mi * 16 + nl;
#pragma unroll
            for (int hp = 0; hp < 8; ++hp) {
                const int K = 2 * hp + (g << 4);
                au[mi].u[hp] =
                    *reinterpret_cast<const unsigned int*>(&lds_a[buf][row][K]);
            }
        }

#pragma unroll
        for (int mi = 0; mi < 2; ++mi)
#pragma unroll
            for (int nt = 0; nt < 4; ++nt)
                acc[mi][nt] = __builtin_amdgcn_wmma_f32_16x16x32_bf16(
                    false, au[mi].v, false, bu[nt].v,
                    (short)0, acc[mi][nt], false, false);
    }

    // ---- epilogue: bias + relu + residual, write fp32 + bf16 ----
#pragma unroll
    for (int mi = 0; mi < 2; ++mi)
#pragma unroll
        for (int nt = 0; nt < 4; ++nt) {
            const int n = n0 + wn * 64 + nt * 16 + nl;
            const float bv = bias[n];
#pragma unroll
            for (int r = 0; r < 8; ++r) {
                const int m = m0 + wm * 32 + mi * 16 + (g << 3) + r;
                float v = acc[mi][nt][r] + bv;
                v = v > 0.f ? v : 0.f;
                const size_t off = (size_t)m * D_CH + n;
                const float o = v + xf[off];
                yf[off] = o;
                yh[off] = f2bf(o);
            }
        }
}

// ---------------------------------------------------------------------------
// LayerNorm + l2norm, row per wave
// ---------------------------------------------------------------------------
__global__ __launch_bounds__(256) void k_ln(const float* __restrict__ x,
                                            const float* __restrict__ sc,
                                            const float* __restrict__ bi,
                                            float* __restrict__ en) {
    const int lane = threadIdx.x & 31;
    const int row  = blockIdx.x * 8 + (threadIdx.x >> 5);
    const float* xr = x + (size_t)row * D_CH;
    float v[16];
    float s = 0.f, sq = 0.f;
#pragma unroll
    for (int j = 0; j < 16; ++j) {
        v[j] = xr[lane + 32 * j];
        s  += v[j];
        sq += v[j] * v[j];
    }
#pragma unroll
    for (int o = 16; o; o >>= 1) { s += __shfl_xor(s, o); sq += __shfl_xor(sq, o); }
    const float mu   = s * (1.f / D_CH);
    const float var  = sq * (1.f / D_CH) - mu * mu;
    const float rstd = rsqrtf(var + 1e-5f);
    float nsq = 0.f;
#pragma unroll
    for (int j = 0; j < 16; ++j) {
        const int d = lane + 32 * j;
        v[j] = (v[j] - mu) * rstd * sc[d] + bi[d];
        nsq += v[j] * v[j];
    }
#pragma unroll
    for (int o = 16; o; o >>= 1) nsq += __shfl_xor(nsq, o);
    const float inv = 1.f / fmaxf(sqrtf(nsq), 1e-12f);
    float* er = en + (size_t)row * D_CH;
#pragma unroll
    for (int j = 0; j < 16; ++j) er[lane + 32 * j] = v[j] * inv;
}

// ---------------------------------------------------------------------------
// Normalize both codebooks, row per wave (rows 0..127 -> z, 128..191 -> q)
// ---------------------------------------------------------------------------
__global__ __launch_bounds__(256) void k_cbnorm(const float* __restrict__ cz,
                                                const float* __restrict__ cq,
                                                float* __restrict__ czn,
                                                float* __restrict__ cqn) {
    const int lane = threadIdx.x & 31;
    const int row  = blockIdx.x * 8 + (threadIdx.x >> 5);
    const float* src;
    float* dst;
    if (row < KZ_N) { src = cz + (size_t)row * D_CH;          dst = czn + (size_t)row * D_CH; }
    else            { src = cq + (size_t)(row - KZ_N) * D_CH; dst = cqn + (size_t)(row - KZ_N) * D_CH; }
    float v[16];
    float sq = 0.f;
#pragma unroll
    for (int j = 0; j < 16; ++j) { v[j] = src[lane + 32 * j]; sq += v[j] * v[j]; }
#pragma unroll
    for (int o = 16; o; o >>= 1) sq += __shfl_xor(sq, o);
    const float inv = 1.f / fmaxf(sqrtf(sq), 1e-12f);
#pragma unroll
    for (int j = 0; j < 16; ++j) dst[lane + 32 * j] = v[j] * inv;
}

// ---------------------------------------------------------------------------
// Level-2 VQ lookup table: per iz -> best iq, and ||z-q||^2 = 2 - 2*sim
// ---------------------------------------------------------------------------
__global__ void k_lut(const float* __restrict__ czn, const float* __restrict__ cqn,
                      int* __restrict__ liq, float* __restrict__ ld2) {
    const int iz = threadIdx.x;               // blockDim == 128
    const float* z = czn + (size_t)iz * D_CH;
    float best = -1e30f;
    int bk = 0;
    for (int k = 0; k < KQ_N; ++k) {
        const float* q = cqn + (size_t)k * D_CH;
        float d = 0.f;
        for (int j = 0; j < D_CH; ++j) d += z[j] * q[j];
        if (d > best) { best = d; bk = k; }
    }
    liq[iz] = bk;
    ld2[iz] = 2.f - 2.f * best;
}

// ---------------------------------------------------------------------------
// VQ level 1 argmax (cosine), level 2 via LUT, per-row loss, decoder input
// ---------------------------------------------------------------------------
__global__ __launch_bounds__(256) void k_vq(const float* __restrict__ en,
                                            const float* __restrict__ czn,
                                            const float* __restrict__ cqn,
                                            const int* __restrict__ liq,
                                            const float* __restrict__ ld2,
                                            float* __restrict__ oiz,
                                            float* __restrict__ oiq,
                                            float* __restrict__ rowloss,
                                            float* __restrict__ qf,
                                            unsigned short* __restrict__ qh) {
    __shared__ float se[8][D_CH];
    const int lane = threadIdx.x & 31;
    const int w    = threadIdx.x >> 5;
    const int row  = blockIdx.x * 8 + w;
    const float* er = en + (size_t)row * D_CH;
#pragma unroll
    for (int j = 0; j < 16; ++j) se[w][lane + 32 * j] = er[lane + 32 * j];
    __syncthreads();

    float best = -1e30f;
    int bk = 0;
    for (int kk = 0; kk < 4; ++kk) {
        const int k = lane + 32 * kk;
        const float* c = czn + (size_t)k * D_CH;
        float d = 0.f;
        for (int j = 0; j < D_CH; ++j) d += se[w][j] * c[j];
        if (d > best || (d == best && k < bk)) { best = d; bk = k; }
    }
#pragma unroll
    for (int o = 16; o; o >>= 1) {
        const float b2 = __shfl_xor(best, o);
        const int   k2 = __shfl_xor(bk, o);
        if (b2 > best || (b2 == best && k2 < bk)) { best = b2; bk = k2; }
    }
    const int iq = liq[bk];
    if (lane == 0) {
        oiz[row]     = (float)bk;
        oiq[row]     = (float)iq;
        rowloss[row] = 1.0f * (2.f - 2.f * best) + 0.8f * ld2[bk];
    }
    const float* q = cqn + (size_t)iq * D_CH;
    float* qfr = qf + (size_t)row * D_CH;
    unsigned short* qhr = qh + (size_t)row * D_CH;
#pragma unroll
    for (int j = 0; j < 16; ++j) {
        const int d = lane + 32 * j;
        const float v = q[d];
        qfr[d] = v;
        qhr[d] = f2bf(v);
    }
}

// ---------------------------------------------------------------------------
// Deterministic loss reduction
// ---------------------------------------------------------------------------
__global__ void k_loss(const float* __restrict__ rowloss, float* __restrict__ out) {
    __shared__ float sm[256];
    float s = 0.f;
    for (int i = threadIdx.x; i < M_TOT; i += 256) s += rowloss[i];
    sm[threadIdx.x] = s;
    __syncthreads();
    for (int o = 128; o; o >>= 1) {
        if (threadIdx.x < (unsigned)o) sm[threadIdx.x] += sm[threadIdx.x + o];
        __syncthreads();
    }
    if (threadIdx.x == 0) out[0] = sm[0] / ((float)M_TOT * (float)D_CH);
}

// ---------------------------------------------------------------------------
// Output projection: [M,512] @ [512,14] + b -> recon
// ---------------------------------------------------------------------------
__global__ __launch_bounds__(256) void k_proj_out(const float* __restrict__ y,
                                                  const float* __restrict__ W,
                                                  const float* __restrict__ b,
                                                  float* __restrict__ out) {
    const int idx = blockIdx.x * 256 + threadIdx.x;
    if (idx >= M_TOT * A_DIM) return;
    const int m = idx / A_DIM;
    const int a = idx - m * A_DIM;
    const float* yr = y + (size_t)m * D_CH;
    float acc = b[a];
    for (int d = 0; d < D_CH; ++d) acc += yr[d] * W[d * A_DIM + a];
    out[idx] = acc;
}

// ---------------------------------------------------------------------------
extern "C" void kernel_launch(void* const* d_in, const int* in_sizes, int n_in,
                              void* d_out, int out_size, void* d_ws, size_t ws_size,
                              hipStream_t stream) {
    const float* actions    = (const float*)d_in[0];
    const float* enc_proj_w = (const float*)d_in[1];
    const float* enc_proj_b = (const float*)d_in[2];
    const float* enc_conv_w = (const float*)d_in[3];
    const float* enc_conv_b = (const float*)d_in[4];
    const float* ln_scale   = (const float*)d_in[5];
    const float* ln_bias    = (const float*)d_in[6];
    const float* cbz        = (const float*)d_in[7];
    const float* cbq        = (const float*)d_in[8];
    const float* dec_conv_w = (const float*)d_in[9];
    const float* dec_conv_b = (const float*)d_in[10];
    const float* dec_proj_w = (const float*)d_in[11];
    const float* dec_proj_b = (const float*)d_in[12];
    float* out = (float*)d_out;

    char* ws = (char*)d_ws;
    unsigned short* WENC = (unsigned short*)(ws + 0);          // 31,457,280 B
    unsigned short* WDEC = WENC + (size_t)WPERL * NLAY;        // 31,457,280 B
    float*          XF   = (float*)(ws + 62914560);            // 33,554,432 B
    float*          YF   = (float*)(ws + 96468992);            // 33,554,432 B
    unsigned short* XH   = (unsigned short*)(ws + 130023424);  // 16,777,216 B
    unsigned short* YH   = (unsigned short*)(ws + 146800640);  // 16,777,216 B
    float*          CZN  = (float*)(ws + 163577856);           //    262,144 B
    float*          CQN  = (float*)(ws + 163840000);           //    131,072 B
    int*            LIQ  = (int*)  (ws + 163971072);           //        512 B
    float*          LD2  = (float*)(ws + 163971584);           //        512 B
    float*          RL   = (float*)(ws + 163972096);           //     65,536 B

    // weight fragments (enc + dec, bf16, WMMA B-layout)
    k_wprep<<<122880, 256, 0, stream>>>(enc_conv_w, dec_conv_w, WENC);
    // codebooks
    k_cbnorm<<<24, 256, 0, stream>>>(cbz, cbq, CZN, CQN);
    k_lut<<<1, 128, 0, stream>>>(CZN, CQN, LIQ, LD2);
    // encoder projection
    k_proj_in<<<32768, 256, 0, stream>>>(actions, enc_proj_w, enc_proj_b, XF, XH);

    float* pf = XF; unsigned short* ph = XH;
    float* qf = YF; unsigned short* qh = YH;
    for (int l = 0; l < NLAY; ++l) {
        k_tcn<<<dim3(128, 4), 256, 0, stream>>>(pf, ph, qf, qh,
            WENC + (size_t)l * WPERL, enc_conv_b + l * D_CH, 1 << (l % 10));
        float* tf = pf; pf = qf; qf = tf;
        unsigned short* th = ph; ph = qh; qh = th;
    }
    // pf == XF again (20 swaps). LayerNorm + l2norm -> e_n in YF.
    k_ln<<<2048, 256, 0, stream>>>(pf, ln_scale, ln_bias, qf);
    // VQ: indices + per-row loss + decoder input (q_st rows) into XF/XH
    k_vq<<<2048, 256, 0, stream>>>(qf, CZN, CQN, LIQ, LD2,
                                   out + 229377, out + 245761, RL, pf, ph);
    k_loss<<<1, 256, 0, stream>>>(RL, out + 229376);

    for (int l = 0; l < NLAY; ++l) {
        k_tcn<<<dim3(128, 4), 256, 0, stream>>>(pf, ph, qf, qh,
            WDEC + (size_t)l * WPERL, dec_conv_b + l * D_CH, 1 << (l % 10));
        float* tf = pf; pf = qf; qf = tf;
        unsigned short* th = ph; ph = qh; qh = th;
    }
    // pf == XF again; final projection -> recon at out[0..229376)
    k_proj_out<<<896, 256, 0, stream>>>(pf, dec_proj_w, dec_proj_b, out);
}
